// RGAT_28312424415233
// MI455X (gfx1250) — matrix operand order repaired and checked by reading, not compiled, not run on hardware
//
#include <hip/hip_runtime.h>
#include <hip/hip_bf16.h>

typedef __attribute__((ext_vector_type(16))) __bf16 v16bf;
typedef __attribute__((ext_vector_type(8)))  __bf16 bf16x8;
typedef __attribute__((ext_vector_type(8)))  float  v8f;

union Frag { v16bf v; bf16x8 h[2]; };

#define APITCH 136   // 128 + 8 bf16 pad -> 4-bank skew per row, conflict-free ds_load_b128

// ---------------- GEMM: H[n, j] = sum_k X[n,k] * W[k,j]   (128x128 weights) -------------
// bf16-split (hi/lo) with fp32 accumulate: ~fp32 accuracy on the bf16 WMMA pipe.
__global__ __launch_bounds__(256) void rgat_gemm(const float* __restrict__ X,
                                                 const float* __restrict__ W,
                                                 float* __restrict__ H, int nrows) {
  __shared__ __align__(16) __bf16 sh_hi[16 * APITCH];
  __shared__ __align__(16) __bf16 sh_lo[16 * APITCH];
  const int t  = threadIdx.x;
  const int r0 = blockIdx.x * 16;

  { // stage 16x128 X tile as bf16 hi/lo
    const int trow = t >> 4;
    const int tcol = (t & 15) << 3;
    int gr = r0 + trow; if (gr >= nrows) gr = nrows - 1;
    const float4* p = (const float4*)(X + (size_t)gr * 128 + tcol);
    float4 f0 = p[0], f1 = p[1];
    float tmp[8] = {f0.x, f0.y, f0.z, f0.w, f1.x, f1.y, f1.z, f1.w};
#pragma unroll
    for (int j = 0; j < 8; ++j) {
      __bf16 hi = (__bf16)tmp[j];
      sh_hi[trow * APITCH + tcol + j] = hi;
      sh_lo[trow * APITCH + tcol + j] = (__bf16)(tmp[j] - (float)hi);
    }
  }
  __syncthreads();

  const int wave = t >> 5;
  const int lane = t & 31;
  const int n0   = wave << 4;            // 16-column tile per wave
  const int arow = lane & 15;
  const int sel8 = (lane >> 4) << 3;     // 0 for lanes 0-15, 8 for lanes 16-31

  v8f c = {0.f, 0.f, 0.f, 0.f, 0.f, 0.f, 0.f, 0.f};

#pragma unroll
  for (int k0 = 0; k0 < 128; k0 += 32) {
    // A fragment (16x32 bf16): lane<16 holds K {k0..k0+7, k0+16..k0+23}; lane>=16 offset +8
    Frag ahi, alo;
    const int base = arow * APITCH + k0 + sel8;
    ahi.h[0] = *(const bf16x8*)&sh_hi[base];
    ahi.h[1] = *(const bf16x8*)&sh_hi[base + 16];
    alo.h[0] = *(const bf16x8*)&sh_lo[base];
    alo.h[1] = *(const bf16x8*)&sh_lo[base + 16];

    // B fragment (32x16 bf16): column n0+(lane&15); element e = K k0+sel8+(e&7)+16*(e>>3)
    const float* wp = W + (size_t)(k0 + sel8) * 128 + n0 + (lane & 15);
    float wt[16];
#pragma unroll
    for (int e = 0; e < 16; ++e) wt[e] = wp[(size_t)((e & 7) + ((e >> 3) << 4)) * 128];
    Frag bhi, blo;
#pragma unroll
    for (int e = 0; e < 16; ++e) {
      __bf16 hi = (__bf16)wt[e];
      bhi.v[e] = hi;
      blo.v[e] = (__bf16)(wt[e] - (float)hi);
    }

    c = __builtin_amdgcn_wmma_f32_16x16x32_bf16(false, ahi.v, false, bhi.v, (short)0, c, false, false);
    c = __builtin_amdgcn_wmma_f32_16x16x32_bf16(false, alo.v, false, bhi.v, (short)0, c, false, false);
    c = __builtin_amdgcn_wmma_f32_16x16x32_bf16(false, ahi.v, false, blo.v, (short)0, c, false, false);
  }

  // C/D layout: VGPR i -> row i (lanes 0-15) / row i+8 (lanes 16-31), N = lane&15
  const int n  = n0 + (lane & 15);
  const int mb = r0 + sel8;
#pragma unroll
  for (int i = 0; i < 8; ++i) {
    int m = mb + i;
    if (m < nrows) H[(size_t)m * 128 + n] = c[i];
  }
}

// -------- order-preserving float<->uint encoding for atomic segment-max --------
__device__ __forceinline__ unsigned encf(float f) {
  unsigned u = __float_as_uint(f);
  return (u & 0x80000000u) ? ~u : (u | 0x80000000u);
}
__device__ __forceinline__ float decf(unsigned u) {
  u = (u & 0x80000000u) ? (u ^ 0x80000000u) : ~u;
  return __uint_as_float(u);
}
__device__ __forceinline__ float lrelu(float v) { return v >= 0.f ? v : 0.2f * v; }

// ------------- per-node attention coefficients el/er, and reset mx/sum ----------------
__global__ __launch_bounds__(256) void attn_coef(const float* __restrict__ H,
                                                 const float* __restrict__ al,
                                                 const float* __restrict__ ar,
                                                 float* __restrict__ el, float* __restrict__ er,
                                                 unsigned* __restrict__ mxu,
                                                 float* __restrict__ sm, int n_nodes) {
  const int node = blockIdx.x * 8 + (threadIdx.x >> 5);
  if (node >= n_nodes) return;
  const int lane = threadIdx.x & 31;                     // lanes 0-15: head0, 16-31: head1
  float4 h4 = ((const float4*)(H + (size_t)node * 128))[lane];
  float4 a4 = ((const float4*)al)[lane];
  float4 r4 = ((const float4*)ar)[lane];
  float se = h4.x * a4.x + h4.y * a4.y + h4.z * a4.z + h4.w * a4.w;
  float sr = h4.x * r4.x + h4.y * r4.y + h4.z * r4.z + h4.w * r4.w;
#pragma unroll
  for (int m = 1; m < 16; m <<= 1) {
    se += __shfl_xor(se, m, 32);
    sr += __shfl_xor(sr, m, 32);
  }
  if ((lane & 15) == 0) {
    int idx = node * 2 + (lane >> 4);
    el[idx] = se; er[idx] = sr; mxu[idx] = 0u; sm[idx] = 0.f;
  }
}

// ------------- edge pass 1: segment max of leaky_relu(el[src]+er[dst]) ----------------
__global__ void edge_max(const int* __restrict__ src, const int* __restrict__ dst,
                         const float* __restrict__ el, const float* __restrict__ er,
                         unsigned* __restrict__ mxu, int n_edges) {
  int e = blockIdx.x * blockDim.x + threadIdx.x;
  if (e >= n_edges) return;
  int s = src[e], d = dst[e];
#pragma unroll
  for (int h = 0; h < 2; ++h) {
    float v = lrelu(el[2 * s + h] + er[2 * d + h]);
    atomicMax(&mxu[2 * d + h], encf(v));
  }
}

// ------------- edge pass 2: softmax denominator -----------------
__global__ void edge_sum(const int* __restrict__ src, const int* __restrict__ dst,
                         const float* __restrict__ el, const float* __restrict__ er,
                         const unsigned* __restrict__ mxu, float* __restrict__ sm, int n_edges) {
  int e = blockIdx.x * blockDim.x + threadIdx.x;
  if (e >= n_edges) return;
  int s = src[e], d = dst[e];
#pragma unroll
  for (int h = 0; h < 2; ++h) {
    float v = lrelu(el[2 * s + h] + er[2 * d + h]);
    atomicAdd(&sm[2 * d + h], expf(v - decf(mxu[2 * d + h])));
  }
}

// ------------- edge pass 3: gather h[src], scale by alpha, scatter-add to acc[dst] ----
__global__ __launch_bounds__(256) void edge_msg(const int* __restrict__ src,
                                                const int* __restrict__ dst,
                                                const float* __restrict__ el,
                                                const float* __restrict__ er,
                                                const unsigned* __restrict__ mxu,
                                                const float* __restrict__ sm,
                                                const float* __restrict__ H,
                                                float* __restrict__ acc, int n_edges) {
  const int e = blockIdx.x * 8 + (threadIdx.x >> 5);
  if (e >= n_edges) return;
  const int lane = threadIdx.x & 31;
  const int s = src[e], d = dst[e];
  const int hd = lane >> 4;                       // head per half-wave
  float v = lrelu(el[2 * s + hd] + er[2 * d + hd]);
  float alpha = expf(v - decf(mxu[2 * d + hd])) / sm[2 * d + hd];
  float4 hv = ((const float4*)(H + (size_t)s * 128))[lane];   // coalesced 512B gather
  float* ap = acc + (size_t)d * 128 + 4 * lane;
  atomicAdd(ap + 0, alpha * hv.x);
  atomicAdd(ap + 1, alpha * hv.y);
  atomicAdd(ap + 2, alpha * hv.z);
  atomicAdd(ap + 3, alpha * hv.w);
}

// ------------- epilogues -----------------
__global__ void init_bias(float* __restrict__ acc, const float* __restrict__ b, int total) {
  int i = blockIdx.x * blockDim.x + threadIdx.x;
  if (i >= total) return;
  int j = i & 127;
  acc[i] = b[j] + b[128 + j] + b[256 + j];       // sum of per-relation biases
}

__global__ void relu_swap(const float* __restrict__ acc, float* __restrict__ h1,
                          const float* __restrict__ b2, float* __restrict__ acc_out, int total) {
  int i = blockIdx.x * blockDim.x + threadIdx.x;
  if (i >= total) return;
  float v = acc[i];
  h1[i] = v > 0.f ? v : 0.f;
  int j = i & 127;
  acc_out[i] = b2[j] + b2[128 + j] + b2[256 + j];
}

__global__ void head_mean(const float* __restrict__ acc, float* __restrict__ out, int n_nodes) {
  int i = blockIdx.x * blockDim.x + threadIdx.x;
  if (i >= n_nodes * 64) return;
  int n = i >> 6, d = i & 63;
  out[i] = 0.5f * (acc[(size_t)n * 128 + d] + acc[(size_t)n * 128 + 64 + d]);
}

extern "C" void kernel_launch(void* const* d_in, const int* in_sizes, int n_in,
                              void* d_out, int out_size, void* d_ws, size_t ws_size,
                              hipStream_t stream) {
  const float* x   = (const float*)d_in[0];
  const int*   src = (const int*)d_in[1];
  const int*   dst = (const int*)d_in[2];
  const float* Wl[2]  = {(const float*)d_in[3], (const float*)d_in[7]};
  const float* all_[2] = {(const float*)d_in[4], (const float*)d_in[8]};
  const float* arl[2] = {(const float*)d_in[5], (const float*)d_in[9]};
  const float* bl[2]  = {(const float*)d_in[6], (const float*)d_in[10]};

  const int n_nodes = in_sizes[0] / 128;
  const int n_edges = in_sizes[1] / 3;
  const size_t NF = (size_t)n_nodes * 128;

  float*    ws   = (float*)d_ws;
  float*    hbuf = ws;                 // [N,128] per-relation features
  float*    acc  = hbuf + NF;         // [N,128] layer accumulator
  float*    h1   = acc + NF;          // [N,128] relu'd layer-1 output
  float*    el   = h1 + NF;           // [N,2]
  float*    er   = el + 2 * (size_t)n_nodes;
  float*    sm   = er + 2 * (size_t)n_nodes;
  unsigned* mxu  = (unsigned*)(sm + 2 * (size_t)n_nodes);

  const int gemm_grid  = (n_nodes + 15) / 16;
  const int nodew_grid = (n_nodes + 7) / 8;
  const int edge_grid  = (n_edges + 255) / 256;
  const int edgew_grid = (n_edges + 7) / 8;
  const int nf_grid    = (int)((NF + 255) / 256);

  init_bias<<<nf_grid, 256, 0, stream>>>(acc, bl[0], (int)NF);

  for (int layer = 0; layer < 2; ++layer) {
    const float* in = layer ? h1 : x;
    for (int r = 0; r < 3; ++r) {
      rgat_gemm<<<gemm_grid, 256, 0, stream>>>(in, Wl[layer] + (size_t)r * 128 * 128, hbuf, n_nodes);
      attn_coef<<<nodew_grid, 256, 0, stream>>>(hbuf, all_[layer] + r * 128, arl[layer] + r * 128,
                                                el, er, mxu, sm, n_nodes);
      const int* sr = src + (size_t)r * n_edges;
      const int* dr = dst + (size_t)r * n_edges;
      edge_max<<<edge_grid, 256, 0, stream>>>(sr, dr, el, er, mxu, n_edges);
      edge_sum<<<edge_grid, 256, 0, stream>>>(sr, dr, el, er, mxu, sm, n_edges);
      edge_msg<<<edgew_grid, 256, 0, stream>>>(sr, dr, el, er, mxu, sm, hbuf, acc, n_edges);
    }
    if (layer == 0)
      relu_swap<<<nf_grid, 256, 0, stream>>>(acc, h1, bl[1], acc, (int)NF);
  }

  head_mean<<<(n_nodes * 64 + 255) / 256, 256, 0, stream>>>(acc, (float*)d_out, n_nodes);
}
